// PPOLSTMDIALAgent_84653805404321
// MI455X (gfx1250) — matrix operand order repaired and checked by reading, not compiled, not run on hardware
//
#include <hip/hip_runtime.h>
#include <math.h>

typedef __attribute__((ext_vector_type(16))) _Float16 v16h;
typedef __attribute__((ext_vector_type(8)))  _Float16 v8h;
typedef __attribute__((ext_vector_type(8)))  float    v8f;

#define TT   256
#define BB   512
#define TBR  (TT * BB)
#define HH   128
#define G4   (4 * HH)
#define HIDW 32

__device__ __forceinline__ v16h cat8(v8h lo, v8h hi) {
  v16h r;
#pragma unroll
  for (int i = 0; i < 8; ++i) { r[i] = lo[i]; r[8 + i] = hi[i]; }
  return r;
}

__device__ __forceinline__ float sigf(float x) { return 1.f / (1.f + __expf(-x)); }

// ---------------------------------------------------------------------------
// Pack a row-major fp32 weight [N, Korig] into fp16 [N, Kpad] (zeros past Korig)
// ---------------------------------------------------------------------------
__global__ void k_pack(const float* __restrict__ W, _Float16* __restrict__ out,
                       int N, int Korig, int Kpad) {
  int idx = blockIdx.x * blockDim.x + threadIdx.x;
  if (idx >= N * Kpad) return;
  int n = idx / Kpad, k = idx - n * Kpad;
  out[idx] = (_Float16)((k < Korig) ? W[(size_t)n * Korig + k] : 0.f);
}

// ---------------------------------------------------------------------------
// Generic WMMA GEMM: out[M,N] = relu(A[M,K] @ W^T + bias), out in f16.
// Each wave: one 16-row M tile x NG 16-col N tiles. KT = K/32 steps.
// A fragment: lane holds row (lane&15); halves h -> k = (h&7)+((h>>3)*16)+((lane>>4)*8)
// B fragment: lane holds col (lane&15); halves h -> k = (h&7)+((h>>3)*16)+((lane>>4)*8), same as A
// ---------------------------------------------------------------------------
template <int KT, int NG, bool AF16>
__global__ void k_gemm_relu(const void* __restrict__ Asrc, float ascale, int lda, int aKorig,
                            const _Float16* __restrict__ pW, int kpad,
                            const float* __restrict__ bias,
                            _Float16* __restrict__ out, int ldo, int col_off,
                            int ngroups) {
  const int w    = threadIdx.x >> 5;
  const int lane = threadIdx.x & 31;
  const int gw   = blockIdx.x * (blockDim.x >> 5) + w;
  const int mt   = gw / ngroups;
  const int ng   = gw - mt * ngroups;
  const int mrow = mt * 16 + (lane & 15);
  const int kb   = (lane >> 4) * 8;   // A-fragment k base
  const int n0   = lane & 15;
  const int kbb  = (lane >> 4) * 8;   // B-fragment k base

  v8f acc[NG] = {};
#pragma unroll
  for (int kt = 0; kt < KT; ++kt) {
    v16h a;
    if constexpr (AF16) {
      const _Float16* ap = (const _Float16*)Asrc + (size_t)mrow * lda + kt * 32 + kb;
      a = cat8(*(const v8h*)ap, *(const v8h*)(ap + 16));
    } else {
      const float* ap = (const float*)Asrc + (size_t)mrow * lda;
#pragma unroll
      for (int h = 0; h < 16; ++h) {
        int k = kt * 32 + (h & 7) + ((h >> 3) * 16) + kb;
        a[h] = (_Float16)((k < aKorig) ? ap[k] * ascale : 0.f);
      }
    }
#pragma unroll
    for (int g = 0; g < NG; ++g) {
      int nt = ng * NG + g;
      const _Float16* bp = pW + (size_t)(nt * 16 + n0) * kpad + kt * 32 + kbb;
      v16h b = cat8(*(const v8h*)bp, *(const v8h*)(bp + 16));
      acc[g] = __builtin_amdgcn_wmma_f32_16x16x32_f16(false, a, false, b, (short)0,
                                                      acc[g], false, false);
    }
  }
  const int r0 = (lane >> 4) * 8;
#pragma unroll
  for (int g = 0; g < NG; ++g) {
    int col = (ng * NG + g) * 16 + n0;
    float bv = bias[col];
#pragma unroll
    for (int v = 0; v < 8; ++v) {
      float val = acc[g][v] + bv;
      val = val > 0.f ? val : 0.f;
      out[(size_t)(mt * 16 + r0 + v) * ldo + col_off + col] = (_Float16)val;
    }
  }
}

// ---------------------------------------------------------------------------
// loc/msg encoders + zero padding of LSTM-input columns 24..31
// ---------------------------------------------------------------------------
__global__ void k_locmsg(const float* __restrict__ loc, const float* __restrict__ msg,
                         const int* __restrict__ done,
                         const float* __restrict__ Wl, const float* __restrict__ bl,
                         const float* __restrict__ Wm, const float* __restrict__ bm,
                         _Float16* __restrict__ hid) {
  int r = blockIdx.x * blockDim.x + threadIdx.x;
  if (r >= TBR) return;
  float lx = loc[2 * r] * 0.1f, ly = loc[2 * r + 1] * 0.1f;
  float m = (1.f - (float)done[r]) * msg[r];
  _Float16* h = hid + (size_t)r * HIDW;
#pragma unroll
  for (int j = 0; j < 4; ++j)
    h[16 + j] = (_Float16)(Wl[2 * j] * lx + Wl[2 * j + 1] * ly + bl[j]);
#pragma unroll
  for (int j = 0; j < 4; ++j) {
    float v = Wm[j] * m + bm[j];
    h[20 + j] = (_Float16)(v > 0.f ? v : 0.f);
  }
#pragma unroll
  for (int j = 0; j < 8; ++j) h[24 + j] = (_Float16)0.f;
}

// ---------------------------------------------------------------------------
// Persistent LSTM: the recurrence is batch-row-local, so each block owns 16
// batch rows and runs ALL 256 time steps. h (f16) and c (f32) live in LDS;
// each wave keeps its Wih/Whh B-fragments in registers for the whole loop.
// Block = 256 threads = 8 waves; wave w covers gate columns [w*64, w*64+64).
// ---------------------------------------------------------------------------
__global__ void __launch_bounds__(256, 1)
k_lstm_all(const _Float16* __restrict__ hid,
           const _Float16* __restrict__ pWih,
           const _Float16* __restrict__ pWhh,
           const float* __restrict__ bih, const float* __restrict__ bhh,
           const float* __restrict__ h0, const float* __restrict__ c0,
           const int* __restrict__ done,
           float* __restrict__ hs, float* __restrict__ hT, float* __restrict__ cT) {
  __shared__ float    gl[16][G4 + 8];   // gate pre-activations
  __shared__ float    csh[16][HH];      // cell state (fp32)
  __shared__ _Float16 hsh[16][HH];      // hidden state (f16, feeds A fragments)
  __shared__ float    cbias[G4];        // bih + bhh

  const int w    = threadIdx.x >> 5;
  const int lane = threadIdx.x & 31;
  const int mt   = blockIdx.x;          // 0..31 -> batch rows [mt*16, mt*16+16)
  const int brow = mt * 16 + (lane & 15);
  const int kb   = (lane >> 4) * 8;
  const int n0   = lane & 15;
  const int kbb  = (lane >> 4) * 8;
  const int r0   = (lane >> 4) * 8;

  // ---- init state + combined bias ----
  for (int it = threadIdx.x; it < 16 * HH; it += blockDim.x) {
    int r = it >> 7, j = it & (HH - 1);
    int bg = mt * 16 + r;
    hsh[r][j] = (_Float16)h0[(size_t)bg * HH + j];
    csh[r][j] = c0[(size_t)bg * HH + j];
  }
  for (int j = threadIdx.x; j < G4; j += blockDim.x) cbias[j] = bih[j] + bhh[j];

  // ---- preload weight fragments into registers (held for all 256 steps) ----
  v16h wih_f[4];
  v16h whh_f[4][4];
#pragma unroll
  for (int g = 0; g < 4; ++g) {
    int nt = w * 4 + g;
    const _Float16* bx = pWih + (size_t)(nt * 16 + n0) * 32 + kbb;
    wih_f[g] = cat8(*(const v8h*)bx, *(const v8h*)(bx + 16));
#pragma unroll
    for (int kt = 0; kt < 4; ++kt) {
      const _Float16* bh = pWhh + (size_t)(nt * 16 + n0) * HH + kt * 32 + kbb;
      whh_f[g][kt] = cat8(*(const v8h*)bh, *(const v8h*)(bh + 16));
    }
  }
  __syncthreads();

  const v16h zfrag = {};

#pragma unroll 1
  for (int t = 0; t < TT; ++t) {
    // done mask (exactly 0 or 1): masked value is a select
    const bool live = (done[t * BB + brow] == 0);

    // x-part A fragment from global (K=32, cols 24..31 zero-padded)
    const _Float16* xp = hid + (size_t)(t * BB + brow) * HIDW + kb;
    v16h ax = cat8(*(const v8h*)xp, *(const v8h*)(xp + 16));

    // h-part A fragments from LDS (K=128), masked
    v16h ah[4];
    const _Float16* hp = &hsh[lane & 15][0];
#pragma unroll
    for (int kt = 0; kt < 4; ++kt) {
      const _Float16* p = hp + kt * 32 + kb;
      v16h f = cat8(*(const v8h*)p, *(const v8h*)(p + 16));
      ah[kt] = live ? f : zfrag;
    }

    v8f acc[4] = {};
#pragma unroll
    for (int g = 0; g < 4; ++g) {
      acc[g] = __builtin_amdgcn_wmma_f32_16x16x32_f16(false, ax, false, wih_f[g],
                                                      (short)0, acc[g], false, false);
#pragma unroll
      for (int kt = 0; kt < 4; ++kt)
        acc[g] = __builtin_amdgcn_wmma_f32_16x16x32_f16(false, ah[kt], false, whh_f[g][kt],
                                                        (short)0, acc[g], false, false);
    }
#pragma unroll
    for (int g = 0; g < 4; ++g) {
      int col = (w * 4 + g) * 16 + n0;
#pragma unroll
      for (int v = 0; v < 8; ++v) gl[r0 + v][col] = acc[g][v];
    }
    __syncthreads();

    // cell update: 16 rows x 128 cols
    float* hout = hs + (size_t)t * BB * HH;
    for (int it = threadIdx.x; it < 16 * HH; it += blockDim.x) {
      int r = it >> 7;
      int j = it & (HH - 1);
      int bg = mt * 16 + r;
      float iv = gl[r][j]          + cbias[j];
      float fv = gl[r][HH + j]     + cbias[HH + j];
      float gv = gl[r][2 * HH + j] + cbias[2 * HH + j];
      float ov = gl[r][3 * HH + j] + cbias[3 * HH + j];
      bool rlive = (done[t * BB + bg] == 0);
      float cp = rlive ? csh[r][j] : 0.f;
      float cn = sigf(fv) * cp + sigf(iv) * tanhf(gv);
      float hn = sigf(ov) * tanhf(cn);
      csh[r][j] = cn;
      hsh[r][j] = (_Float16)hn;
      hout[(size_t)bg * HH + j] = hn;
      if (t == TT - 1) {
        hT[(size_t)bg * HH + j] = hn;
        cT[(size_t)bg * HH + j] = cn;
      }
    }
    __syncthreads();  // protect hsh/csh for next step's fragment loads
  }
}

// ---------------------------------------------------------------------------
extern "C" void kernel_launch(void* const* d_in, const int* in_sizes, int n_in,
                              void* d_out, int out_size, void* d_ws, size_t ws_size,
                              hipStream_t stream) {
  (void)in_sizes; (void)n_in; (void)out_size; (void)ws_size;
  const float* image    = (const float*)d_in[0];
  const float* location = (const float*)d_in[1];
  const float* message  = (const float*)d_in[2];
  const int*   done     = (const int*)d_in[3];
  const float* h0       = (const float*)d_in[4];
  const float* c0       = (const float*)d_in[5];
  const float* Wv1 = (const float*)d_in[6];  const float* bv1 = (const float*)d_in[7];
  const float* Wv2 = (const float*)d_in[8];  const float* bv2 = (const float*)d_in[9];
  const float* Wv3 = (const float*)d_in[10]; const float* bv3 = (const float*)d_in[11];
  const float* Wv4 = (const float*)d_in[12]; const float* bv4 = (const float*)d_in[13];
  const float* Wl  = (const float*)d_in[14]; const float* bl  = (const float*)d_in[15];
  const float* Wm  = (const float*)d_in[16]; const float* bm  = (const float*)d_in[17];
  const float* Wih = (const float*)d_in[18]; const float* bih = (const float*)d_in[19];
  const float* Whh = (const float*)d_in[20]; const float* bhh = (const float*)d_in[21];

  char* ws = (char*)d_ws;
  size_t off = 0;
  auto alloc = [&](size_t bytes) -> void* {
    void* p = ws + off;
    off = (off + bytes + 255) & ~(size_t)255;
    return p;
  };
  _Float16* pWv1 = (_Float16*)alloc((size_t)256 * 32 * 2);
  _Float16* pWv2 = (_Float16*)alloc((size_t)256 * 256 * 2);
  _Float16* pWv3 = (_Float16*)alloc((size_t)128 * 256 * 2);
  _Float16* pWv4 = (_Float16*)alloc((size_t)16 * 128 * 2);
  _Float16* pWih = (_Float16*)alloc((size_t)G4 * 32 * 2);
  _Float16* pWhh = (_Float16*)alloc((size_t)G4 * HH * 2);
  _Float16* act_a = (_Float16*)alloc((size_t)TBR * 256 * 2);
  _Float16* act_b = (_Float16*)alloc((size_t)TBR * 256 * 2);
  _Float16* hidb  = (_Float16*)alloc((size_t)TBR * HIDW * 2);

  float* hs = (float*)d_out;                 // [TB, H]
  float* hT = hs + (size_t)TBR * HH;         // [B, H]
  float* cT = hT + (size_t)BB * HH;          // [B, H]

  // Pack weights -> padded fp16
  auto packN = [&](const float* W, _Float16* P, int N, int Ko, int Kp) {
    int tot = N * Kp;
    k_pack<<<(tot + 255) / 256, 256, 0, stream>>>(W, P, N, Ko, Kp);
  };
  packN(Wv1, pWv1, 256, 25, 32);
  packN(Wv2, pWv2, 256, 256, 256);
  packN(Wv3, pWv3, 128, 256, 256);
  packN(Wv4, pWv4, 16, 128, 128);
  packN(Wih, pWih, G4, 24, 32);
  packN(Whh, pWhh, G4, HH, HH);

  const int MT = TBR / 16;  // 8192 M tiles
  // E1: [TB,25] (f32, /255) -> [TB,256]
  k_gemm_relu<1, 4, false><<<(MT * 4) / 8, 256, 0, stream>>>(
      image, 1.f / 255.f, 25, 25, pWv1, 32, bv1, act_a, 256, 0, 4);
  // E2: [TB,256] -> [TB,256]
  k_gemm_relu<8, 4, true><<<(MT * 4) / 8, 256, 0, stream>>>(
      act_a, 1.f, 256, 256, pWv2, 256, bv2, act_b, 256, 0, 4);
  // E3: [TB,256] -> [TB,128]
  k_gemm_relu<8, 4, true><<<(MT * 2) / 8, 256, 0, stream>>>(
      act_b, 1.f, 256, 256, pWv3, 256, bv3, act_a, 128, 0, 2);
  // E4: [TB,128] -> image_feat, into hid cols 0..15
  k_gemm_relu<4, 1, true><<<MT / 8, 256, 0, stream>>>(
      act_a, 1.f, 128, 128, pWv4, 128, bv4, hidb, HIDW, 0, 1);
  // loc/msg feats into hid cols 16..23, zero 24..31
  k_locmsg<<<TBR / 256, 256, 0, stream>>>(location, message, done,
                                          Wl, bl, Wm, bm, hidb);

  // Entire LSTM recurrence in one persistent kernel (batch-row-local).
  k_lstm_all<<<BB / 16, 256, 0, stream>>>(hidb, pWih, pWhh, bih, bhh,
                                          h0, c0, done, hs, hT, cT);
}
